// Attention_29326036697707
// MI455X (gfx1250) — matrix operand order repaired and checked by reading, not compiled, not run on hardware
//
#include <hip/hip_runtime.h>
#include <hip/hip_bf16.h>
#include <math.h>

// Fused causal attention for MI455X (gfx1250, wave32, WMMA).
// 512 blocks (one per batch), 256 threads = 8 waves. Wave w owns query rows
// [16w, 16w+16). x is read from global exactly once and converted to f16 into
// a register-resident A-strip; weights are staged f32->f16 into LDS once per
// block so WMMA B-fragments are raw ds_load_b128 (no per-tile conversions).

typedef __attribute__((ext_vector_type(16))) _Float16 v16h;
typedef __attribute__((ext_vector_type(8)))  _Float16 v8h;
typedef __attribute__((ext_vector_type(4)))  _Float16 v4h;
typedef __attribute__((ext_vector_type(8)))  float    v8f;

#define T_CTX 128
#define C_DIM 512
#define H_DIM 64
#define NWAVE 8

// ---- fragment loaders -------------------------------------------------------
// A-matrix (16-bit, 16x32): lanes 0-15 hold row M=lane, K = {0..7, 16..23};
// lanes 16-31 hold same rows with K offset +8.
__device__ __forceinline__ v16h frag_a_f32(const float* __restrict__ row, int kc, int lane) {
  const int base = kc + ((lane >> 4) << 3);
  const float4 q0 = *(const float4*)(row + base);
  const float4 q1 = *(const float4*)(row + base + 4);
  const float4 q2 = *(const float4*)(row + base + 16);
  const float4 q3 = *(const float4*)(row + base + 20);
  v16h r;
  r[0]  = (_Float16)q0.x; r[1]  = (_Float16)q0.y; r[2]  = (_Float16)q0.z; r[3]  = (_Float16)q0.w;
  r[4]  = (_Float16)q1.x; r[5]  = (_Float16)q1.y; r[6]  = (_Float16)q1.z; r[7]  = (_Float16)q1.w;
  r[8]  = (_Float16)q2.x; r[9]  = (_Float16)q2.y; r[10] = (_Float16)q2.z; r[11] = (_Float16)q2.w;
  r[12] = (_Float16)q3.x; r[13] = (_Float16)q3.y; r[14] = (_Float16)q3.z; r[15] = (_Float16)q3.w;
  return r;
}

__device__ __forceinline__ v16h frag_a_lds(const _Float16* __restrict__ row, int kc, int lane) {
  const int base = kc + ((lane >> 4) << 3);
  const v8h lo = *(const v8h*)(row + base);
  const v8h hi = *(const v8h*)(row + base + 16);
  v16h r;
#pragma unroll
  for (int i = 0; i < 8; ++i) { r[i] = lo[i]; r[i + 8] = hi[i]; }
  return r;
}

// B-matrix (16-bit, 32x16): lane holds column N=lane&15; lanes 0-15 K={0..7,16..23},
// lanes 16-31 K offset +8 (matches A) -> two ds_load_b128 per fragment.
__device__ __forceinline__ v16h frag_b_lds(const _Float16* __restrict__ row, int kc, int lane) {
  const int base = kc + ((lane >> 4) << 3);
  const v8h lo = *(const v8h*)(row + base);
  const v8h hi = *(const v8h*)(row + base + 16);
  v16h r;
#pragma unroll
  for (int i = 0; i < 8; ++i) { r[i] = lo[i]; r[i + 8] = hi[i]; }
  return r;
}

#define WMMA_F16(a, b, c) \
  __builtin_amdgcn_wmma_f32_16x16x32_f16(false, (a), false, (b), (short)0, (c), false, false)

// LDS layout (144 KB dynamic shared, 2 blocks/WGP within the 320 KB pool):
//   Qh  : [128][64]  f16 @ 0       (16 KB)
//   Kh  : [128][64]  f16 @ 16384   (16 KB)
//   Vt  : [64][128]  f16 @ 32768   (16 KB, transposed V)
//   Whm : [64][512]  f16 @ 49152   (64 KB, phase-1 per-matrix weight stage)
//   S   : [128][128] f32 @ 49152   (64 KB, phase-2, aliases Whm)
//   P   : [128][128] f16 @ 114688  (32 KB, phase-2)
#define SMEM_BYTES 147456

__global__ __launch_bounds__(256) void attn_fused_kernel(
    const float* __restrict__ x,  const float* __restrict__ Wq,
    const float* __restrict__ Wk, const float* __restrict__ Wv,
    float* __restrict__ out)
{
  extern __shared__ char smem[];
  _Float16* Qh  = (_Float16*)(smem);
  _Float16* Kh  = (_Float16*)(smem + 16384);
  _Float16* Vt  = (_Float16*)(smem + 32768);
  _Float16* Whm = (_Float16*)(smem + 49152);
  float*    S   = (float*)   (smem + 49152);
  _Float16* P   = (_Float16*)(smem + 114688);

  const int lane = threadIdx.x & 31;
  const int wave = threadIdx.x >> 5;   // == M-tile owned by this wave
  const int b    = blockIdx.x;
  const float* xb = x + (size_t)b * T_CTX * C_DIM;

  // ---------------- Phase 1: Q/K/V projections (x @ W^T) --------------------
  // Register-resident A-strip: this wave's 16 query rows, all K=512, as f16.
  // 16 fragments x 8 VGPRs = 128 VGPRs, loaded/converted exactly once.
  const float* arow = xb + (size_t)(wave * 16 + (lane & 15)) * C_DIM;
  v16h astrip[16];
#pragma unroll
  for (int k = 0; k < 16; ++k) astrip[k] = frag_a_f32(arow, k * 32, lane);

  for (int mat = 0; mat < 3; ++mat) {
    const float* Wsel = (mat == 0) ? Wq : (mat == 1) ? Wk : Wv;
    __syncthreads();  // previous matrix's B-fragment reads done before overwrite
    // Cooperative stage: W[mat] (64x512 f32, L2-resident) -> f16 in LDS.
    for (int idx = threadIdx.x; idx < (H_DIM * C_DIM) / 4; idx += 256) {
      const int e = idx * 4;
      const float4 q = *(const float4*)(Wsel + e);
      v4h hh;
      hh[0] = (_Float16)q.x; hh[1] = (_Float16)q.y;
      hh[2] = (_Float16)q.z; hh[3] = (_Float16)q.w;
      *(v4h*)(Whm + e) = hh;
    }
    __syncthreads();

    for (int ntile = 0; ntile < 4; ++ntile) {
      const _Float16* brow = Whm + (ntile * 16 + (lane & 15)) * C_DIM;
      v8f acc = {};
#pragma unroll
      for (int k = 0; k < 16; ++k) {
        const v16h bf = frag_b_lds(brow, k * 32, lane);
        acc = WMMA_F16(astrip[k], bf, acc);
      }
      const int n     = lane & 15;
      const int mbase = wave * 16 + ((lane >> 4) << 3);
      if (mat == 0) {
#pragma unroll
        for (int j = 0; j < 8; ++j)
          Qh[(mbase + j) * H_DIM + ntile * 16 + n] = (_Float16)acc[j];
      } else if (mat == 1) {
#pragma unroll
        for (int j = 0; j < 8; ++j)
          Kh[(mbase + j) * H_DIM + ntile * 16 + n] = (_Float16)acc[j];
      } else { // V stored transposed: Vt[h][t]
#pragma unroll
        for (int j = 0; j < 8; ++j)
          Vt[(ntile * 16 + n) * T_CTX + (mbase + j)] = (_Float16)acc[j];
      }
    }
  }
  __syncthreads();  // Qh/Kh/Vt complete; Whm reads done before S overwrites

  // ---------------- Phase 2: scores, causal softmax, P @ V ------------------
  const int w = wave;

  // scores S = (Q K^T) / sqrt(64), causal-masked; only lower-tri tiles
  for (int nt = 0; nt <= w; ++nt) {
    const _Float16* qrow = Qh + (w  * 16 + (lane & 15)) * H_DIM;
    const _Float16* krow = Kh + (nt * 16 + (lane & 15)) * H_DIM;
    v8f acc = {};
#pragma unroll
    for (int kc = 0; kc < H_DIM; kc += 32) {
      const v16h a  = frag_a_lds(qrow, kc, lane);
      const v16h bf = frag_b_lds(krow, kc, lane);
      acc = WMMA_F16(a, bf, acc);
    }
    const int n     = lane & 15;
    const int scol  = nt * 16 + n;
    const int mbase = w * 16 + ((lane >> 4) << 3);
#pragma unroll
    for (int j = 0; j < 8; ++j) {
      const int t = mbase + j;
      S[t * T_CTX + scol] = (scol <= t) ? acc[j] * 0.125f : -__builtin_inff();
    }
  }

  // softmax over cols [0, ncols); P zero-padded to a multiple of 32 columns
  const int ncols = (w + 1) * 16;
  const int kpad  = (ncols + 31) & ~31;
  for (int r = 0; r < 16; ++r) {
    const int t = w * 16 + r;
    float vals[4];
    float m = -__builtin_inff();
#pragma unroll
    for (int i = 0; i < 4; ++i) {
      const int c = lane + 32 * i;
      vals[i] = (c < ncols) ? S[t * T_CTX + c] : -__builtin_inff();
      m = fmaxf(m, vals[i]);
    }
#pragma unroll
    for (int off = 16; off > 0; off >>= 1) m = fmaxf(m, __shfl_xor(m, off, 32));
    float e[4];
    float sum = 0.f;
#pragma unroll
    for (int i = 0; i < 4; ++i) {
      const int c = lane + 32 * i;
      e[i] = (c < ncols) ? __expf(vals[i] - m) : 0.f;
      sum += e[i];
    }
#pragma unroll
    for (int off = 16; off > 0; off >>= 1) sum += __shfl_xor(sum, off, 32);
    const float inv = 1.0f / sum;
#pragma unroll
    for (int i = 0; i < 4; ++i) {
      const int c = lane + 32 * i;
      if (c < kpad)
        P[t * T_CTX + c] = (c < ncols) ? (_Float16)(e[i] * inv) : (_Float16)0.f;
    }
  }

  // out = P @ V (K runs only over the un-masked, zero-padded prefix)
  for (int nt = 0; nt < 4; ++nt) {
    const _Float16* prow = P  + (w  * 16 + (lane & 15)) * T_CTX;
    const _Float16* vrow = Vt + (nt * 16 + (lane & 15)) * T_CTX;
    v8f acc = {};
    for (int kc = 0; kc < kpad; kc += 32) {
      const v16h a  = frag_a_lds(prow, kc, lane);
      const v16h bf = frag_b_lds(vrow, kc, lane);
      acc = WMMA_F16(a, bf, acc);
    }
    const int n     = lane & 15;
    const int mbase = w * 16 + ((lane >> 4) << 3);
#pragma unroll
    for (int j = 0; j < 8; ++j)
      out[((size_t)b * T_CTX + (mbase + j)) * H_DIM + nt * 16 + n] = acc[j];
  }
}

extern "C" void kernel_launch(void* const* d_in, const int* in_sizes, int n_in,
                              void* d_out, int out_size, void* d_ws, size_t ws_size,
                              hipStream_t stream) {
  (void)in_sizes; (void)n_in; (void)out_size; (void)d_ws; (void)ws_size;
  const float* x  = (const float*)d_in[0];
  const float* Wq = (const float*)d_in[1];
  const float* Wk = (const float*)d_in[2];
  const float* Wv = (const float*)d_in[3];
  float* out = (float*)d_out;

  // 144 KB dynamic LDS (> default 64 KB cap); WGP supports up to 320 KB.
  // Idempotent + deterministic, safe under graph capture (not a stream op).
  hipFuncSetAttribute((const void*)attn_fused_kernel,
                      hipFuncAttributeMaxDynamicSharedMemorySize, SMEM_BYTES);

  attn_fused_kernel<<<512, 256, SMEM_BYTES, stream>>>(x, Wq, Wk, Wv, out);
}
